// Attention_35287451304934
// MI455X (gfx1250) — matrix operand order repaired and checked by reading, not compiled, not run on hardware
//
#include <hip/hip_runtime.h>
#include <hip/hip_bf16.h>

typedef __attribute__((ext_vector_type(16))) _Float16 v16h;
typedef __attribute__((ext_vector_type(8)))  float    v8f;

#define B_   4
#define C_   32
#define D_   64
#define H_   64
#define W_   64
#define HW_  (H_ * W_)
#define DHW_ (D_ * HW_)
#define KCONV 686          // 2 * 7*7*7 (true reduction length)
#define QTOT  98           // 2 * 7*7   (ic,dz,dy rows)
#define KP2   800          // k = q*8 + dx (dx padded 7->8), 784 padded to 25*32
#define WINW  70           // 64 + 2*3 padded width
#define WINSZ (QTOT * WINW)        // 6860 halfs of real window data
#define RSTRIDE (WINSZ + 20)       // + zeroed tail so dx=7 edge reads stay in-bounds

// ---------------------------------------------------------------------------
// Kernel 0: pack conv1 weights into zero-padded f16 panel Wb[16][KP2] with the
// dx-padded K order: k = q*8 + dx, q = ic*49 + dz*7 + dy. dx==7 / q>=98 / n>=4
// are zero so the WMMA N/K padding contributes nothing.
// ---------------------------------------------------------------------------
__global__ void prep_weights(const float* __restrict__ w1, _Float16* __restrict__ wb) {
    int t = blockIdx.x * 256 + threadIdx.x;          // 16*KP2 = 12800 threads
    if (t >= 16 * KP2) return;
    int n = t / KP2, k = t - n * KP2;
    int dx = k & 7, q = k >> 3;
    float v = 0.f;
    if (n < 4 && dx < 7 && q < QTOT) v = w1[n * KCONV + q * 7 + dx];
    wb[t] = (_Float16)v;
}

// ---------------------------------------------------------------------------
// Kernel 1: channel pooling: avg & max over D*H*W per (b,c).
// ---------------------------------------------------------------------------
__global__ __launch_bounds__(256) void chan_pool(const float* __restrict__ x,
                                                 float* __restrict__ pavg,
                                                 float* __restrict__ pmax) {
    __shared__ float ssum[256];
    __shared__ float smax[256];
    int bc = blockIdx.x;                              // 0..127
    const float4* xp = (const float4*)(x + (size_t)bc * DHW_);
    float s = 0.f, m = -3.4e38f;
    for (int i = threadIdx.x; i < DHW_ / 4; i += 256) {
        float4 v = xp[i];
        s += v.x + v.y + v.z + v.w;
        m = fmaxf(m, fmaxf(fmaxf(v.x, v.y), fmaxf(v.z, v.w)));
    }
    ssum[threadIdx.x] = s; smax[threadIdx.x] = m;
    __syncthreads();
    for (int off = 128; off > 0; off >>= 1) {
        if (threadIdx.x < off) {
            ssum[threadIdx.x] += ssum[threadIdx.x + off];
            smax[threadIdx.x] = fmaxf(smax[threadIdx.x], smax[threadIdx.x + off]);
        }
        __syncthreads();
    }
    if (threadIdx.x == 0) {
        pavg[bc] = ssum[0] * (1.f / (float)DHW_);
        pmax[bc] = smax[0];
    }
}

// ---------------------------------------------------------------------------
// Kernel 2: channel-attention MLP (tiny). One block of 128 threads.
// ---------------------------------------------------------------------------
__global__ __launch_bounds__(128) void channel_fc(const float* __restrict__ pavg,
                                                  const float* __restrict__ pmax,
                                                  const float* __restrict__ fc1w,
                                                  const float* __restrict__ fc1b,
                                                  const float* __restrict__ fc2w,
                                                  const float* __restrict__ fc2b,
                                                  float* __restrict__ ca) {
    __shared__ float hb[B_][128];
    int j = threadIdx.x;                              // 0..127 hidden unit
    for (int b = 0; b < B_; ++b) {
        float acc = fc1b[j];
        #pragma unroll 8
        for (int i = 0; i < 32; ++i) acc += fc1w[j * 64 + i]      * pavg[b * 32 + i];
        #pragma unroll 8
        for (int i = 0; i < 32; ++i) acc += fc1w[j * 64 + 32 + i] * pmax[b * 32 + i];
        hb[b][j] = fmaxf(acc, 0.f);
    }
    __syncthreads();
    int b = j >> 5, c = j & 31;
    float acc = fc2b[c];
    #pragma unroll 8
    for (int k = 0; k < 128; ++k) acc += fc2w[c * 128 + k] * hb[b][k];
    ca[j] = 1.f / (1.f + __expf(-acc));
}

// ---------------------------------------------------------------------------
// Kernel 3: spatial pooling: mean & max over C per position, stored f16.
// s16 layout: [B][2][DHW]  (plane 0 = avg, 1 = max)
// ---------------------------------------------------------------------------
__global__ __launch_bounds__(256) void spat_pool(const float* __restrict__ x,
                                                 _Float16* __restrict__ s16) {
    int gid = blockIdx.x * 256 + threadIdx.x;         // 0 .. B*DHW-1 (exact)
    int b = gid / DHW_, p = gid - b * DHW_;
    const float* xp = x + (size_t)b * C_ * DHW_ + p;
    float s = 0.f, m = -3.4e38f;
    #pragma unroll 8
    for (int c = 0; c < C_; ++c) {
        float v = xp[(size_t)c * DHW_];
        s += v; m = fmaxf(m, v);
    }
    s16[(size_t)(b * 2 + 0) * DHW_ + p] = (_Float16)(s * (1.f / (float)C_));
    s16[(size_t)(b * 2 + 1) * DHW_ + p] = (_Float16)m;
}

// ---------------------------------------------------------------------------
// Kernel 4: spatial attention conv via WMMA implicit GEMM.
//   M=16 w-positions, N=16 (4 out-ch used), K=800 -> 25 v_wmma_f32_16x16x32_f16.
//   dx-padded K order makes every lane's 8-element fragment run contiguous in
//   the LDS window: pure immediate-offset ds_load_u16, no address VALU chain.
//   Epilogue: relu -> 1x1 conv2 -> sigmoid -> sa[b][dhw].
// One workgroup (8 waves) = 2 (b,d,h) rows x 4 w-tiles of 16.
// ---------------------------------------------------------------------------
__global__ __launch_bounds__(256) void conv_sa(const _Float16* __restrict__ s16,
                                               const _Float16* __restrict__ wb16,
                                               const float* __restrict__ conv2w,
                                               float* __restrict__ sa) {
    __shared__ _Float16 raw[2][RSTRIDE];    // per-row window rows: [q=(ic,dz,dy)][wi]
    __shared__ float    tileD[8][16][4];    // per-wave D scatter (M x outch)

    int tid  = threadIdx.x;
    int wave = tid >> 5, lane = tid & 31;
    int row0 = blockIdx.x * 2;              // global row = b*4096 + d*64 + h

    // keep the weight panel hot (L2-resident, reused by every workgroup)
    __builtin_prefetch((const char*)wb16 + (size_t)tid * 100, 0, 3);

    // ---- cooperative window staging: waves 0-3 -> row0, waves 4-7 -> row0+1.
    // Each wave owns a contiguous q-range; per q-row decode once, lanes sweep w.
    {
        int wv  = wave & 3;
        int lr  = wave >> 2;
        int row = row0 + lr;
        int b  = row >> 12;
        int d0 = (row >> 6) & 63;
        int h0 = row & 63;
        int q0 = wv * 25;
        int q1 = (wv == 3) ? QTOT : q0 + 25;
        for (int q = q0; q < q1; ++q) {
            int ic = (q >= 49) ? 1 : 0;
            int r  = q - ic * 49;
            int dz = r / 7;
            int dy = r - dz * 7;
            int d = d0 + dz - 3, h = h0 + dy - 3;
            bool okdh = (d >= 0) && (d < D_) && (h >= 0) && (h < H_);
            const _Float16* src = s16 + (size_t)(b * 2 + ic) * DHW_ + d * HW_ + h * W_;
            _Float16* dst = &raw[lr][q * WINW];
            for (int wi = lane; wi < WINW; wi += 32) {
                int w = wi - 3;
                _Float16 v = (_Float16)0.f;
                if (okdh && w >= 0 && w < W_) v = src[w];
                dst[wi] = v;
            }
        }
        // zero the tail pad (keeps dx==7 edge reads finite & in-bounds)
        if (tid < 2 * (RSTRIDE - WINSZ)) {
            int lrp = tid / (RSTRIDE - WINSZ);
            int ip  = tid - lrp * (RSTRIDE - WINSZ);
            raw[lrp][WINSZ + ip] = (_Float16)0.f;
        }
    }
    __syncthreads();

    // ---- per-wave WMMA tile: 16 consecutive w outputs x 16 out channels
    int wrow  = wave >> 2;
    int wtile = wave & 3;
    int m     = lane & 15;                   // A row = output position
    int hi    = lane >> 4;                   // fragment half (ISA A/B K split)
    int w0    = wtile * 16;
    int nB    = lane & 15;                   // B column = out channel

    // lane base: element e<8 -> rwl[kk*280 + e], e>=8 -> rwl[kk*280 + 140 + e-8]
    const _Float16* rwl = &raw[wrow][0] + hi * WINW + w0 + m;
    const _Float16* wp  = wb16 + (size_t)nB * KP2 + hi * 8;

    v8f c = {};
    #pragma unroll 4
    for (int kk = 0; kk < 24; ++kk) {
        const _Float16* rb = rwl + kk * (4 * WINW);
        v16h a;
        #pragma unroll
        for (int e = 0; e < 8; ++e) a[e]     = rb[e];
        #pragma unroll
        for (int e = 0; e < 8; ++e) a[8 + e] = rb[2 * WINW + e];
        v16h bf;
        #pragma unroll
        for (int e = 0; e < 8; ++e) { bf[e] = wp[kk * 32 + e]; bf[8 + e] = wp[kk * 32 + 16 + e]; }
        c = __builtin_amdgcn_wmma_f32_16x16x32_f16(false, a, false, bf,
                                                   (short)0, c, false, false);
    }
    {   // kk = 24: k 768..783 valid (first half-run); k 784..799 are pure padding
        const _Float16* rb = rwl + 24 * (4 * WINW);
        v16h a;
        #pragma unroll
        for (int e = 0; e < 8; ++e) a[e]     = rb[e];
        #pragma unroll
        for (int e = 0; e < 8; ++e) a[8 + e] = (_Float16)0.f;
        v16h bf;
        #pragma unroll
        for (int e = 0; e < 8; ++e) { bf[e] = wp[24 * 32 + e]; bf[8 + e] = wp[24 * 32 + 16 + e]; }
        c = __builtin_amdgcn_wmma_f32_16x16x32_f16(false, a, false, bf,
                                                   (short)0, c, false, false);
    }

    // ---- D layout: lane -> N column (lane&15), element e -> M = e + hi*8
    if (nB < 4) {
        int mofs = hi * 8;
        #pragma unroll
        for (int e = 0; e < 8; ++e) tileD[wave][mofs + e][nB] = c[e];
    }
    __syncthreads();

    if (lane < 16) {
        float pre = 0.f;
        #pragma unroll
        for (int oc = 0; oc < 4; ++oc)
            pre += conv2w[oc] * fmaxf(tileD[wave][lane][oc], 0.f);
        float sav = 1.f / (1.f + __expf(-pre));
        int row = row0 + wrow;
        int b = row >> 12, dh = row & 4095;
        sa[(size_t)b * DHW_ + dh * 64 + w0 + lane] = sav;
    }
}

// ---------------------------------------------------------------------------
// Kernel 5: attention = sa * ca (broadcast), anti = 1 - attention. float4 I/O.
// ---------------------------------------------------------------------------
__global__ __launch_bounds__(256) void fuse_out(const float* __restrict__ sa,
                                                const float* __restrict__ ca,
                                                float4* __restrict__ out) {
    size_t v = (size_t)blockIdx.x * 256 + threadIdx.x;   // over B*C*DHW/4 (exact)
    const size_t PV = DHW_ / 4;
    int    bc = (int)(v / PV);
    size_t pv = v - (size_t)bc * PV;
    int    b  = bc >> 5;
    float4 s4 = ((const float4*)sa)[(size_t)b * PV + pv];
    float  cv = ca[bc];
    float4 att;
    att.x = s4.x * cv; att.y = s4.y * cv; att.z = s4.z * cv; att.w = s4.w * cv;
    out[v] = att;
    float4 anti;
    anti.x = 1.f - att.x; anti.y = 1.f - att.y; anti.z = 1.f - att.z; anti.w = 1.f - att.w;
    out[(size_t)(B_ * C_) * PV + v] = anti;
}

// ---------------------------------------------------------------------------
extern "C" void kernel_launch(void* const* d_in, const int* in_sizes, int n_in,
                              void* d_out, int out_size, void* d_ws, size_t ws_size,
                              hipStream_t stream) {
    const float* x      = (const float*)d_in[0];
    const float* fc1w   = (const float*)d_in[1];
    const float* fc1b   = (const float*)d_in[2];
    const float* fc2w   = (const float*)d_in[3];
    const float* fc2b   = (const float*)d_in[4];
    const float* conv1w = (const float*)d_in[5];
    const float* conv2w = (const float*)d_in[6];

    char* ws = (char*)d_ws;
    float*    pavg = (float*)(ws + 0);                      // 128 f
    float*    pmax = (float*)(ws + 512);                    // 128 f
    float*    ca   = (float*)(ws + 1024);                   // 128 f
    _Float16* wb16 = (_Float16*)(ws + 2048);                // 16*800 halfs = 25600 B
    float*    sa   = (float*)(ws + 32768);                  // B*DHW f = 4 MiB
    _Float16* s16  = (_Float16*)(ws + 32768 + 4194304);     // B*2*DHW halfs = 4 MiB

    prep_weights<<<50, 256, 0, stream>>>(conv1w, wb16);                 // 12800 thr
    chan_pool  <<<128, 256, 0, stream>>>(x, pavg, pmax);                // 128 (b,c)
    channel_fc <<<1, 128, 0, stream>>>(pavg, pmax, fc1w, fc1b, fc2w, fc2b, ca);
    spat_pool  <<<(B_ * DHW_) / 256, 256, 0, stream>>>(x, s16);         // 4096 blk
    conv_sa    <<<(B_ * D_ * H_) / 2, 256, 0, stream>>>(s16, wb16, conv2w, sa); // 8192 blk
    fuse_out   <<<(B_ * C_ * DHW_ / 4) / 256, 256, 0, stream>>>(sa, ca, (float4*)d_out); // 32768 blk
}
